// GaussianBlock_81621558493688
// MI455X (gfx1250) — matrix operand order repaired and checked by reading, not compiled, not run on hardware
//
#include <hip/hip_runtime.h>

// ---------------------------------------------------------------------------
// GaussianBlock for MI455X (gfx1250): bf16 WMMA GEMMs + flash attention.
// B=4, L=1024, H=768, NH=12, dh=64, K=3 branches, FFN=3072.
// GEMM-shaped work -> v_wmma_f32_16x16x32_bf16 (fp32 accum); B-weight tiles
// staged to LDS via the Tensor Data Mover (tensor_load_to_lds) when available.
// ---------------------------------------------------------------------------

#define HD    768
#define SEQ   1024
#define BATCH 4
#define NHEAD 12
#define DHEAD 64
#define KBR   3
#define FFD   3072

typedef __attribute__((ext_vector_type(16))) __bf16 v16bf;
typedef __attribute__((ext_vector_type(2)))  __bf16 v2bf;
typedef __attribute__((ext_vector_type(8)))  float  v8f;
typedef __attribute__((ext_vector_type(4)))  unsigned int u32x4;
typedef __attribute__((ext_vector_type(8)))  int i32x8;
typedef __attribute__((ext_vector_type(4)))  int i32x4;

#ifndef __has_builtin
#define __has_builtin(x) 0
#endif
#if __has_builtin(__builtin_amdgcn_tensor_load_to_lds) && \
    __has_builtin(__builtin_amdgcn_s_wait_tensorcnt)
#define HAVE_TDM 1
#else
#define HAVE_TDM 0
#endif

#if HAVE_TDM
#if __clang_major__ >= 23
#define TDM_LOAD(g0, g1, g2, g3)                                     \
  do {                                                               \
    i32x8 _z8 = {0, 0, 0, 0, 0, 0, 0, 0};                            \
    __builtin_amdgcn_tensor_load_to_lds(g0, g1, g2, g3, _z8, 0);     \
  } while (0)
#else
#define TDM_LOAD(g0, g1, g2, g3)                                     \
  __builtin_amdgcn_tensor_load_to_lds(g0, g1, g2, g3, 0)
#endif
#endif

union FragBF { v16bf v; unsigned int u[8]; };

// native f32->bf16 (RNE) conversions; clang lowers fptrunc-to-bf16 to the
// hardware cvt on gfx1250 (v_cvt_pk_bf16_f32 when paired).
__device__ __forceinline__ unsigned int pack2bf(float lo, float hi) {
  v2bf t;
  t.x = (__bf16)lo;
  t.y = (__bf16)hi;
  return __builtin_bit_cast(unsigned int, t);
}
__device__ __forceinline__ unsigned short f2bf(float x) {
  return __builtin_bit_cast(unsigned short, (__bf16)x);
}
__device__ __forceinline__ v8f zero8() {
  v8f z = {0.f,0.f,0.f,0.f,0.f,0.f,0.f,0.f};
  return z;
}
__device__ __forceinline__ v8f wmma_bf16(const FragBF& a, const FragBF& b, v8f c) {
  // (neg_a, A, neg_b, B, c_mod, C, reuse_a, reuse_b)
  return __builtin_amdgcn_wmma_f32_16x16x32_bf16(false, a.v, false, b.v,
                                                 (short)0, c, false, false);
}

// ---------------------------------------------------------------------------
// Generic WMMA GEMM:  C[M,N] = act(A[M,Kd] @ W[Kd,N] + bias[N])
// Block = 128 threads (4 waves), tile 64x64, K-step 32.
// Kd, N multiples of 32/64 for all call sites; M bounded.
// B tile (32x64 f32) staged to LDS by the TDM; A tile staged as bf16.
// ---------------------------------------------------------------------------
__global__ __launch_bounds__(128) void gemm_wmma(const float* __restrict__ A,
                                                 const float* __restrict__ Bw,
                                                 const float* __restrict__ bias,
                                                 float* __restrict__ C,
                                                 int M, int N, int Kd, int relu)
{
  __shared__ unsigned short As[64][32];    // A tile (bf16), [m][k]
  __shared__ float Bsf[32][64];            // B tile (f32),  [k][n]

  const int tid  = threadIdx.x;
  const int lane = tid & 31;
  const int wave = tid >> 5;
  const int rowBase = blockIdx.y * 64;
  const int colBase = blockIdx.x * 64;
  const int wm = (wave >> 1) * 32;
  const int wn = (wave & 1) * 32;

  v8f acc[2][2];
  #pragma unroll
  for (int i = 0; i < 2; ++i)
    #pragma unroll
    for (int j = 0; j < 2; ++j) acc[i][j] = zero8();

  const int arow  = tid >> 1;        // 0..63
  const int acseg = (tid & 1) * 16;
  const int brow  = tid >> 2;        // 0..31
  const int bcseg = (tid & 3) * 16;

  for (int k0 = 0; k0 < Kd; k0 += 32) {
    // ---- stage A tile (bf16, row-major) ----
    {
      int gr = rowBase + arow;
      if (gr < M) {
        const float4* ap =
            reinterpret_cast<const float4*>(A + (size_t)gr * Kd + k0 + acseg);
        #pragma unroll
        for (int q = 0; q < 4; ++q) {
          float4 t = ap[q];
          *reinterpret_cast<unsigned int*>(&As[arow][acseg + 4*q])     = pack2bf(t.x, t.y);
          *reinterpret_cast<unsigned int*>(&As[arow][acseg + 4*q + 2]) = pack2bf(t.z, t.w);
        }
        if (k0 + 32 < Kd)
          __builtin_prefetch(A + (size_t)gr * Kd + k0 + 32 + acseg, 0, 0);
      } else {
        #pragma unroll
        for (int q = 0; q < 16; ++q) As[arow][acseg + q] = 0;
      }
    }

    // ---- stage B tile (f32) ----
#if HAVE_TDM
    if (wave == 0) {
      // Tensor DMA descriptor (D#), groups 0/1 per CDNA5 ISA ch.8.
      unsigned long long ga =
          (unsigned long long)(const void*)(Bw + (size_t)k0 * N + colBase);
      unsigned int ldsa = (unsigned int)(unsigned long long)(void*)&Bsf[0][0];
      u32x4 g0 = { 1u,                                  // count=1 (valid D#)
                   ldsa,                                // lds_addr
                   (unsigned int)(ga & 0xFFFFFFFFu),    // global_addr[31:0]
                   (unsigned int)((ga >> 32) & 0x01FFFFFFu) | (2u << 30) }; // type=2
      unsigned int td0 = (unsigned int)(N - colBase);   // tensor_dim0 (elems)
      unsigned int td1 = (unsigned int)(Kd - k0);       // tensor_dim1 (rows)
      i32x8 g1;
      g1[0] = 0x00020000;                               // data_size=2 (4 bytes)
      g1[1] = (int)((td0 & 0xFFFFu) << 16);             // tensor_dim0 lo16
      g1[2] = (int)(((td1 & 0xFFFFu) << 16) | (td0 >> 16));
      g1[3] = (int)((64u << 16) | (td1 >> 16));         // tile_dim0=64
      g1[4] = 32;                                       // tile_dim1=32
      g1[5] = (int)(unsigned int)N;                     // tensor_dim0_stride
      g1[6] = 0;
      g1[7] = 0;
      i32x4 gz = {0, 0, 0, 0};
      TDM_LOAD(g0, g1, gz, gz);
      __builtin_amdgcn_s_wait_tensorcnt(0);
    }
#else
    {
      const float4* bp = reinterpret_cast<const float4*>(
          Bw + (size_t)(k0 + brow) * N + colBase + bcseg);
      #pragma unroll
      for (int q = 0; q < 4; ++q)
        *reinterpret_cast<float4*>(&Bsf[brow][bcseg + 4*q]) = bp[q];
      if (k0 + 32 < Kd)
        __builtin_prefetch(Bw + (size_t)(k0 + 32 + brow) * N + colBase + bcseg, 0, 0);
    }
#endif
    __syncthreads();

    // ---- build fragments (layouts per ISA 7.12.2, wave32) ----
    const int l15 = lane & 15;
    const int kbA = (lane < 16) ? 0 : 8;    // A: 16x32, half-wave K base
    const int kbB = (lane < 16) ? 0 : 16;   // unused: B fragments also use kbA
    FragBF af[2], bf[2];
    #pragma unroll
    for (int mi = 0; mi < 2; ++mi) {
      int r = wm + mi * 16 + l15;
      #pragma unroll
      for (int v = 0; v < 8; ++v) {
        int kk = ((v & 4) ? 16 : 0) + kbA + 2 * (v & 3);
        af[mi].u[v] = *reinterpret_cast<const unsigned int*>(&As[r][kk]);
      }
    }
    #pragma unroll
    for (int ni = 0; ni < 2; ++ni) {
      int c = wn + ni * 16 + l15;
      #pragma unroll
      for (int v = 0; v < 8; ++v) {
        int kk = ((v & 4) ? 16 : 0) + kbA + 2 * (v & 3);
        bf[ni].u[v] = pack2bf(Bsf[kk][c], Bsf[kk + 1][c]);
      }
    }
    #pragma unroll
    for (int mi = 0; mi < 2; ++mi)
      #pragma unroll
      for (int ni = 0; ni < 2; ++ni)
        acc[mi][ni] = wmma_bf16(af[mi], bf[ni], acc[mi][ni]);

    __syncthreads();
  }

  // ---- epilogue: C layout VGPR r -> row (r + 8*half), col = lane&15 ----
  const int cl15 = lane & 15;
  const int half = lane >> 4;
  #pragma unroll
  for (int mi = 0; mi < 2; ++mi) {
    #pragma unroll
    for (int ni = 0; ni < 2; ++ni) {
      int gc = colBase + wn + ni * 16 + cl15;
      #pragma unroll
      for (int r = 0; r < 8; ++r) {
        int gr = rowBase + wm + mi * 16 + r + 8 * half;
        if (gr < M && gc < N) {
          float val = acc[mi][ni][r];
          if (bias) val += bias[gc];
          if (relu) val = fmaxf(val, 0.f);
          C[(size_t)gr * N + gc] = val;
        }
      }
    }
  }
}

// ---------------------------------------------------------------------------
// Flash attention, one wave per (batch, head, 16-query tile).
// width <= 0 -> no Gaussian bias; else bias = max((dmin^2-d^2)/(2wt^2), ln 1e-6)
// ---------------------------------------------------------------------------
__global__ __launch_bounds__(32) void flash_attn(const float* __restrict__ Q,
                                                 const float* __restrict__ Km,
                                                 const float* __restrict__ Vm,
                                                 const float* __restrict__ mask,
                                                 float* __restrict__ O,
                                                 float width)
{
  __shared__ unsigned short Pl[16][32];   // probability tile (bf16)

  const int lane = threadIdx.x;
  const int qt   = blockIdx.x;            // query tile (0..63)
  const int h    = blockIdx.y;
  const int b    = blockIdx.z;
  const int l15  = lane & 15;
  const int half = lane >> 4;
  const size_t base = ((size_t)b * SEQ) * HD + (size_t)h * DHEAD;

  // Q fragments: A layout 16x32, two dh-chunks
  FragBF qf[2];
  {
    const float* qp = Q + base + (size_t)(qt * 16 + l15) * HD;
    const int kb = half * 8;
    #pragma unroll
    for (int c = 0; c < 2; ++c)
      #pragma unroll
      for (int v = 0; v < 8; ++v) {
        int d0 = c * 32 + ((v & 4) ? 16 : 0) + kb + 2 * (v & 3);
        float2 t = *reinterpret_cast<const float2*>(qp + d0);
        qf[c].u[v] = pack2bf(t.x, t.y);
      }
  }

  float mrow[8], lrow[8];
  #pragma unroll
  for (int r = 0; r < 8; ++r) { mrow[r] = -1e30f; lrow[r] = 0.f; }
  v8f oacc[4];
  #pragma unroll
  for (int c = 0; c < 4; ++c) oacc[c] = zero8();

  const int useBias = (width > 0.f);
  float inv2wt2 = 0.f;
  if (useBias) {
    float wt = fmaxf(width, 0.01f) / 5.0f;   // gauss_divisor = 5
    inv2wt2 = 1.0f / (2.0f * wt * wt);
  }
  const float LOGMIN = -13.815511f;          // ln(1e-6)

  for (int kb0 = 0; kb0 < SEQ; kb0 += 32) {
    // ---- S = Q @ K^T : two 16x16 key sub-tiles, contraction over dh=64 ----
    v8f s[2];
    #pragma unroll
    for (int j = 0; j < 2; ++j) {
      s[j] = zero8();
      const float* kp = Km + base + (size_t)(kb0 + j * 16 + l15) * HD;
      const int kh = half * 16;
      #pragma unroll
      for (int c = 0; c < 2; ++c) {
        FragBF kf;
        #pragma unroll
        for (int v = 0; v < 8; ++v) {
          int d0 = c * 32 + ((v & 4) ? 16 : 0) + half * 8 + 2 * (v & 3);
          float2 t = *reinterpret_cast<const float2*>(kp + d0);
          kf.u[v] = pack2bf(t.x, t.y);
        }
        s[j] = wmma_bf16(qf[c], kf, s[j]);
      }
    }

    const float mk0 = mask[(size_t)b * SEQ + kb0 + l15];
    const float mk1 = mask[(size_t)b * SEQ + kb0 + 16 + l15];

    float p0[8], p1[8];
    #pragma unroll
    for (int r = 0; r < 8; ++r) {
      const int qg = qt * 16 + r + 8 * half;
      float s0 = s[0][r] * 0.125f + (1.f - mk0) * -10000.f;
      float s1 = s[1][r] * 0.125f + (1.f - mk1) * -10000.f;
      if (useBias) {
        float center = (float)qg * (1.0f / (float)SEQ);
        float kstar  = floorf(center * (float)(SEQ - 1) + 0.5f);
        float dmin   = kstar * (1.0f / (float)(SEQ - 1)) - center;
        float dminsq = dmin * dmin;
        int   key0   = kb0 + l15;
        float d0 = (float)key0        * (1.0f / (float)(SEQ - 1)) - center;
        float d1 = (float)(key0 + 16) * (1.0f / (float)(SEQ - 1)) - center;
        s0 += fmaxf((dminsq - d0 * d0) * inv2wt2, LOGMIN);
        s1 += fmaxf((dminsq - d1 * d1) * inv2wt2, LOGMIN);
      }
      // row reductions across the 16-lane half-wave (xor < 16 stays in half)
      float loc = fmaxf(s0, s1);
      loc = fmaxf(loc, __shfl_xor(loc, 1, 32));
      loc = fmaxf(loc, __shfl_xor(loc, 2, 32));
      loc = fmaxf(loc, __shfl_xor(loc, 4, 32));
      loc = fmaxf(loc, __shfl_xor(loc, 8, 32));
      float mnew  = fmaxf(mrow[r], loc);
      float scale = __expf(mrow[r] - mnew);
      mrow[r] = mnew;
      float e0 = __expf(s0 - mnew);
      float e1 = __expf(s1 - mnew);
      float rs = e0 + e1;
      rs += __shfl_xor(rs, 1, 32);
      rs += __shfl_xor(rs, 2, 32);
      rs += __shfl_xor(rs, 4, 32);
      rs += __shfl_xor(rs, 8, 32);
      lrow[r] = lrow[r] * scale + rs;
      #pragma unroll
      for (int c = 0; c < 4; ++c) oacc[c][r] *= scale;
      p0[r] = e0; p1[r] = e1;
    }

    // ---- C-layout -> A-layout for P via LDS ----
    __syncthreads();
    #pragma unroll
    for (int r = 0; r < 8; ++r) {
      int row = r + 8 * half;
      Pl[row][l15]      = f2bf(p0[r]);
      Pl[row][16 + l15] = f2bf(p1[r]);
    }
    __syncthreads();

    FragBF pf;
    {
      const int kb = half * 8;
      #pragma unroll
      for (int v = 0; v < 8; ++v) {
        int kk = ((v & 4) ? 16 : 0) + kb + 2 * (v & 3);
        pf.u[v] = *reinterpret_cast<const unsigned int*>(&Pl[l15][kk]);
      }
    }

    // ---- O += P @ V : 4 dh-chunks of 16 cols, contraction over 32 keys ----
    const int kh = half * 16;
    #pragma unroll
    for (int c = 0; c < 4; ++c) {
      FragBF vf;
      #pragma unroll
      for (int v = 0; v < 8; ++v) {
        int key = kb0 + ((v & 4) ? 16 : 0) + half * 8 + 2 * (v & 3);
        float a0 = Vm[base + (size_t)key       * HD + c * 16 + l15];
        float a1 = Vm[base + (size_t)(key + 1) * HD + c * 16 + l15];
        vf.u[v] = pack2bf(a0, a1);
      }
      oacc[c] = wmma_bf16(pf, vf, oacc[c]);
    }
    __syncthreads();
  }

  #pragma unroll
  for (int c = 0; c < 4; ++c)
    #pragma unroll
    for (int r = 0; r < 8; ++r) {
      int qg = qt * 16 + r + 8 * half;
      O[base + (size_t)qg * HD + c * 16 + l15] = oacc[c][r] / lrow[r];
    }
}

// ---------------------------------------------------------------------------
// y = LayerNorm(x + res) * gamma + beta     (res may be nullptr)
// ---------------------------------------------------------------------------
__global__ __launch_bounds__(256) void add_layernorm(const float* __restrict__ x,
                                                     const float* __restrict__ res,
                                                     const float* __restrict__ gamma,
                                                     const float* __restrict__ beta,
                                                     float* __restrict__ y,
                                                     int rows)
{
  __shared__ float red[256];
  const int row = blockIdx.x;
  if (row >= rows) return;
  const int t = threadIdx.x;
  const float* xr = x + (size_t)row * HD;
  const float* rr = res ? res + (size_t)row * HD : nullptr;
  float v[3];
  #pragma unroll
  for (int i = 0; i < 3; ++i) {
    int c = t + i * 256;
    v[i] = xr[c] + (rr ? rr[c] : 0.f);
  }
  red[t] = v[0] + v[1] + v[2];
  __syncthreads();
  for (int off = 128; off > 0; off >>= 1) {
    if (t < off) red[t] += red[t + off];
    __syncthreads();
  }
  const float mean = red[0] * (1.0f / (float)HD);
  __syncthreads();
  float d0 = v[0] - mean, d1 = v[1] - mean, d2 = v[2] - mean;
  red[t] = d0 * d0 + d1 * d1 + d2 * d2;
  __syncthreads();
  for (int off = 128; off > 0; off >>= 1) {
    if (t < off) red[t] += red[t + off];
    __syncthreads();
  }
  const float rstd = rsqrtf(red[0] * (1.0f / (float)HD) + 1e-5f);
  float dv[3] = {d0, d1, d2};
  #pragma unroll
  for (int i = 0; i < 3; ++i) {
    int c = t + i * 256;
    y[(size_t)row * HD + c] = dv[i] * rstd * gamma[c] + beta[c];
  }
}

// masked mean over L of branch-mean:  sel[b,c] = sum_l m*(oo0+oo1+oo2)/3 / max(sum m,1)
__global__ void mean_sel(const float* __restrict__ oo0, const float* __restrict__ oo1,
                         const float* __restrict__ oo2, const float* __restrict__ mask,
                         float* __restrict__ sel)
{
  const int c = blockIdx.x * 256 + threadIdx.x;   // 0..767
  const int b = blockIdx.y;
  float acc = 0.f, msum = 0.f;
  for (int l = 0; l < SEQ; ++l) {
    float m = mask[(size_t)b * SEQ + l];
    size_t idx = ((size_t)b * SEQ + l) * HD + c;
    acc  += m * (oo0[idx] + oo1[idx] + oo2[idx]) * (1.0f / 3.0f);
    msum += m;
  }
  sel[(size_t)b * HD + c] = acc / fmaxf(msum, 1.f);
}

__global__ void fill_qt(const float* __restrict__ sel, float* __restrict__ qt)
{
  const int idx = blockIdx.x * 256 + threadIdx.x;   // < 12*768
  const int row = idx / HD;
  const int c   = idx - row * HD;
  qt[idx] = sel[(size_t)(row / KBR) * HD + c];
}

// cross-attention: 1 query token over SEQ keys, per (batch, head), one branch
__global__ __launch_bounds__(256) void cross_attn(const float* __restrict__ cq,
                                                  const float* __restrict__ ck,
                                                  const float* __restrict__ cv,
                                                  const float* __restrict__ mask,
                                                  float* __restrict__ out,
                                                  int branch)
{
  __shared__ float qv[DHEAD];
  __shared__ float sc[SEQ];
  __shared__ float red[256];
  const int h  = blockIdx.x;
  const int b  = blockIdx.y;
  const int t  = threadIdx.x;
  const int bk = b * KBR + branch;
  if (t < DHEAD) qv[t] = cq[(size_t)bk * HD + h * DHEAD + t];
  __syncthreads();
  for (int j = t; j < SEQ; j += 256) {
    const float* kp = ck + ((size_t)b * SEQ + j) * HD + h * DHEAD;
    float dot = 0.f;
    #pragma unroll 8
    for (int d = 0; d < DHEAD; ++d) dot += qv[d] * kp[d];
    sc[j] = dot * 0.125f + (1.f - mask[(size_t)b * SEQ + j]) * -10000.f;
  }
  __syncthreads();
  float loc = -1e30f;
  for (int j = t; j < SEQ; j += 256) loc = fmaxf(loc, sc[j]);
  red[t] = loc; __syncthreads();
  for (int off = 128; off > 0; off >>= 1) {
    if (t < off) red[t] = fmaxf(red[t], red[t + off]);
    __syncthreads();
  }
  const float bmax = red[0];
  __syncthreads();
  float ls = 0.f;
  for (int j = t; j < SEQ; j += 256) { float e = __expf(sc[j] - bmax); sc[j] = e; ls += e; }
  red[t] = ls; __syncthreads();
  for (int off = 128; off > 0; off >>= 1) {
    if (t < off) red[t] += red[t + off];
    __syncthreads();
  }
  const float denom = red[0];
  __syncthreads();
  if (t < DHEAD) {
    float o = 0.f;
    for (int j = 0; j < SEQ; ++j)
      o += sc[j] * cv[((size_t)b * SEQ + j) * HD + h * DHEAD + t];
    out[(size_t)bk * HD + h * DHEAD + t] = o / denom;
  }
}

__global__ __launch_bounds__(256) void final_logits(const float* __restrict__ wfeat,
                                                    const float* __restrict__ Wl2,
                                                    const float* __restrict__ bl2,
                                                    float* __restrict__ logits)
{
  __shared__ float red[256];
  const int row = blockIdx.x;
  const int t = threadIdx.x;
  float s = 0.f;
  for (int c = t; c < HD; c += 256) s += wfeat[(size_t)row * HD + c] * Wl2[c];
  red[t] = s; __syncthreads();
  for (int off = 128; off > 0; off >>= 1) {
    if (t < off) red[t] += red[t + off];
    __syncthreads();
  }
  if (t == 0) logits[row] = red[0] + bl2[0];
}

__global__ void softmax_w(const float* __restrict__ logits, float* __restrict__ w)
{
  const int b = threadIdx.x;
  if (b < BATCH) {
    float a0 = logits[b * KBR + 0], a1 = logits[b * KBR + 1], a2 = logits[b * KBR + 2];
    float m = fmaxf(a0, fmaxf(a1, a2));
    float e0 = __expf(a0 - m), e1 = __expf(a1 - m), e2 = __expf(a2 - m);
    float s = e0 + e1 + e2;
    w[b * KBR + 0] = e0 / s; w[b * KBR + 1] = e1 / s; w[b * KBR + 2] = e2 / s;
  }
}

__global__ void combine_out(const float* __restrict__ oo0, const float* __restrict__ oo1,
                            const float* __restrict__ oo2, const float* __restrict__ w,
                            float* __restrict__ out)
{
  const size_t idx = (size_t)blockIdx.x * 256 + threadIdx.x;
  const int b = (int)(idx / ((size_t)SEQ * HD));
  out[idx] = oo0[idx] * w[b * KBR + 0] + oo1[idx] * w[b * KBR + 1]
           + oo2[idx] * w[b * KBR + 2];
}

// ---------------------------------------------------------------------------
extern "C" void kernel_launch(void* const* d_in, const int* in_sizes, int n_in,
                              void* d_out, int out_size, void* d_ws, size_t ws_size,
                              hipStream_t stream)
{
  (void)in_sizes; (void)n_in; (void)out_size; (void)ws_size;

  const float* X    = (const float*)d_in[0];
  const float* msk  = (const float*)d_in[1];
  const float* bWq  = (const float*)d_in[2];
  const float* bWk  = (const float*)d_in[3];
  const float* bWv  = (const float*)d_in[4];
  const float* bW1  = (const float*)d_in[5];
  const float* bW2  = (const float*)d_in[6];
  const float* caWq = (const float*)d_in[7];
  const float* caWk = (const float*)d_in[8];
  const float* caWv = (const float*)d_in[9];
  const float* caW1 = (const float*)d_in[10];
  const float* caW2 = (const float*)d_in[11];
  const float* Wl1  = (const float*)d_in[12];
  const float* Wl2  = (const float*)d_in[13];
  const float* bbq  = (const float*)d_in[14];
  const float* bbk  = (const float*)d_in[15];
  const float* bbv  = (const float*)d_in[16];
  const float* bb1  = (const float*)d_in[17];
  const float* bb2  = (const float*)d_in[18];
  const float* cabq = (const float*)d_in[19];
  const float* cabk = (const float*)d_in[20];
  const float* cabv = (const float*)d_in[21];
  const float* cab1 = (const float*)d_in[22];
  const float* cab2 = (const float*)d_in[23];
  const float* bl1  = (const float*)d_in[24];
  const float* bl2  = (const float*)d_in[25];
  const float* bn1s = (const float*)d_in[26];
  const float* bn2s = (const float*)d_in[27];
  const float* bns  = (const float*)d_in[28];
  const float* can1s= (const float*)d_in[29];
  const float* can2s= (const float*)d_in[30];
  const float* bn1b = (const float*)d_in[31];
  const float* bn2b = (const float*)d_in[32];
  const float* bnb  = (const float*)d_in[33];
  const float* can1b= (const float*)d_in[34];
  const float* can2b= (const float*)d_in[35];

  float* ws = (float*)d_ws;
  const size_t U = (size_t)BATCH * SEQ * HD;   // 3,145,728 floats
  float* Qb  = ws;               // Q proj, then x1
  float* Kb  = ws + 1 * U;
  float* Vb  = ws + 2 * U;
  float* Db  = ws + 3 * U;       // attention output
  float* Eb  = ws + 4 * U;       // FFN hidden (4 units: 4U..8U)
  float* Fb  = ws + 8 * U;       // FFN out
  float* Gb  = ws + 9 * U;       // block out (pre branch-LN)
  float* OO  = ws + 10 * U;      // 3 branch outputs (10U..13U)
  float* CKb = ws + 13 * U;
  float* CVb = ws + 14 * U;
  float* SM  = ws + 15 * U;      // small buffers
  float* sel    = SM;
  float* qt     = SM + 16384;
  float* cq     = SM + 2 * 16384;
  float* ca_out = SM + 3 * 16384;
  float* q1     = SM + 4 * 16384;
  float* t1     = SM + 5 * 16384;
  float* t2     = SM + 6 * 16384;
  float* q2     = SM + 7 * 16384;
  float* wfeat  = SM + 8 * 16384;
  float* logits = SM + 9 * 16384;
  float* wsm    = SM + 9 * 16384 + 64;

  auto gemm = [&](const float* A, const float* W, const float* bias, float* C,
                  int M, int N, int Kd, int relu) {
    dim3 grid((N + 63) / 64, (M + 63) / 64);
    gemm_wmma<<<grid, 128, 0, stream>>>(A, W, bias, C, M, N, Kd, relu);
  };

  const int Mtok = BATCH * SEQ;                       // 4096
  const float widths[KBR] = {-1.f, 2.f, 4.f};         // WIDTHS = (None, 2, 4)

  // ---- K=3 self-attention transformer branches ----
  for (int i = 0; i < KBR; ++i) {
    const size_t oHH = (size_t)i * HD * HD;
    gemm(X, bWq + oHH, bbq + i * HD, Qb, Mtok, HD, HD, 0);
    gemm(X, bWk + oHH, bbk + i * HD, Kb, Mtok, HD, HD, 0);
    gemm(X, bWv + oHH, bbv + i * HD, Vb, Mtok, HD, HD, 0);
    dim3 ag(SEQ / 16, NHEAD, BATCH);
    flash_attn<<<ag, 32, 0, stream>>>(Qb, Kb, Vb, msk, Db, widths[i]);
    add_layernorm<<<Mtok, 256, 0, stream>>>(X, Db, bn1s + i * HD, bn1b + i * HD, Qb, Mtok);
    gemm(Qb, bW1 + (size_t)i * HD * FFD, bb1 + i * FFD, Eb, Mtok, FFD, HD, 1);
    gemm(Eb, bW2 + (size_t)i * FFD * HD, bb2 + i * HD, Fb, Mtok, HD, FFD, 0);
    add_layernorm<<<Mtok, 256, 0, stream>>>(Qb, Fb, bn2s + i * HD, bn2b + i * HD, Gb, Mtok);
    add_layernorm<<<Mtok, 256, 0, stream>>>(Gb, nullptr, bns, bnb, OO + (size_t)i * U, Mtok);
  }

  // ---- fusion head ----
  dim3 mg(HD / 256, BATCH);
  mean_sel<<<mg, 256, 0, stream>>>(OO, OO + U, OO + 2 * U, msk, sel);
  fill_qt<<<(BATCH * KBR * HD) / 256, 256, 0, stream>>>(sel, qt);
  gemm(qt, caWq, cabq, cq, BATCH * KBR, HD, HD, 0);
  for (int i = 0; i < KBR; ++i) {
    gemm(OO + (size_t)i * U, caWk, cabk, CKb, Mtok, HD, HD, 0);
    gemm(OO + (size_t)i * U, caWv, cabv, CVb, Mtok, HD, HD, 0);
    dim3 cg(NHEAD, BATCH);
    cross_attn<<<cg, 256, 0, stream>>>(cq, CKb, CVb, msk, ca_out, i);
  }
  add_layernorm<<<BATCH * KBR, 256, 0, stream>>>(qt, ca_out, can1s, can1b, q1, BATCH * KBR);
  gemm(q1, caW1, cab1, t1, BATCH * KBR, HD, HD, 1);
  gemm(t1, caW2, cab2, t2, BATCH * KBR, HD, HD, 0);
  add_layernorm<<<BATCH * KBR, 256, 0, stream>>>(q1, t2, can2s, can2b, q2, BATCH * KBR);
  gemm(q2, Wl1, bl1, wfeat, BATCH * KBR, HD, HD, 1);
  final_logits<<<BATCH * KBR, 256, 0, stream>>>(wfeat, Wl2, bl2, logits);
  softmax_w<<<1, 32, 0, stream>>>(logits, wsm);
  combine_out<<<(int)(U / 256), 256, 0, stream>>>(OO, OO + U, OO + 2 * U, wsm, (float*)d_out);
}
